// SovereignLeviathanV2_2929167695982
// MI455X (gfx1250) — compile-verified, matrix operand order checked by reading, not run on hardware
//
#include <hip/hip_runtime.h>

// ---------------------------------------------------------------------------
// MoE FFN for MI455X (gfx1250, wave32, WMMA + async global->LDS).
//
// Prep (one-time, memory-bound):
//   x  -> xh/xl        bf16 hi/lo planes, row-major (B,S,D)
//   W1 -> w1h/w1l      bf16 hi/lo planes, TRANSPOSED to (E,F,D)  (N-major)
//   W2 -> w2h/w2l      bf16 hi/lo planes, TRANSPOSED to (E,D,F)  (N-major)
// GEMM1: H(hi/lo planes) = gelu(X @ W1[e] + b1[e])
// GEMM2: Y = (H @ W2[e] + b2[e]) * w[b]
//
// Split-bf16 accuracy:  A*B ~= Ah*Bh + Ah*Bl + Al*Bh  (3 WMMA per K=32 slab,
// fp32 accumulate, ~2^-16 relative error vs fp32 reference).
// Stage fill uses GLOBAL_LOAD_ASYNC_TO_LDS_B128 (ASYNCcnt) — no VALU, no
// register staging, per-lane LDS scatter keeps the padded-row layout.
// ---------------------------------------------------------------------------

typedef __attribute__((ext_vector_type(16))) __bf16        v16bf;
typedef __attribute__((ext_vector_type(8)))  float         v8f;
typedef __attribute__((ext_vector_type(8)))  unsigned int  v8u;

#define TILE_M   128
#define TILE_N   128
#define TILE_K   32
#define LDS_PAD  40                      // 80B row stride: 16B-aligned, 20 banks
#define STAGE_BYTES (TILE_M * LDS_PAD * 2)  // 10240 B per plane per stage

__device__ __forceinline__ void split_bf16(float f, unsigned short& h, unsigned short& l) {
    unsigned int u  = __builtin_bit_cast(unsigned int, f);
    unsigned short hh = (unsigned short)(u >> 16);            // truncate to bf16
    float hf        = __builtin_bit_cast(float, (unsigned int)hh << 16);
    unsigned int ul = __builtin_bit_cast(unsigned int, f - hf);
    h = hh;
    l = (unsigned short)(ul >> 16);                           // residual bf16
}

__device__ __forceinline__ unsigned lds_off(const void* p) {
    return (unsigned)(unsigned long long)p;   // low 32 bits = LDS byte offset
}

// Async 16B-per-lane copy, global -> LDS (ASYNCcnt-tracked).
__device__ __forceinline__ void async_b128(unsigned lds, const unsigned short* g) {
    asm volatile("global_load_async_to_lds_b128 %0, %1, off"
                 :: "v"(lds), "v"((unsigned long long)g)
                 : "memory");
}
__device__ __forceinline__ void wait_async0() {
    asm volatile("s_wait_asynccnt 0" ::: "memory");
}

// 16x32 bf16 WMMA operand fragment from a row-major padded LDS tile.
// Lane layout (ISA 7.12.2): row = lane&15; K-half = (lane>=16)?8:0; two
// contiguous 16B groups at +0/+32 bytes.
__device__ __forceinline__ v16bf load_frag(const unsigned short* base, int row, int khalf) {
    const uint4* p0 = (const uint4*)(base + row * LDS_PAD + khalf);
    const uint4* p1 = (const uint4*)(base + row * LDS_PAD + khalf + 16);
    uint4 a = *p0;
    uint4 b = *p1;
    v8u u = { a.x, a.y, a.z, a.w, b.x, b.y, b.z, b.w };
    return __builtin_bit_cast(v16bf, u);
}

// ---------------------------------------------------------------------------
// Prep kernel 1: elementwise fp32 -> bf16 hi/lo planes (vectorized x4).
// ---------------------------------------------------------------------------
__global__ __launch_bounds__(256)
void split_planes_kernel(const float* __restrict__ in,
                         unsigned short* __restrict__ hi,
                         unsigned short* __restrict__ lo, size_t n4) {
    size_t i = (size_t)blockIdx.x * blockDim.x + threadIdx.x;
    if (i >= n4) return;
    float4 v = ((const float4*)in)[i];
    unsigned short h0, h1, h2, h3, l0, l1, l2, l3;
    split_bf16(v.x, h0, l0); split_bf16(v.y, h1, l1);
    split_bf16(v.z, h2, l2); split_bf16(v.w, h3, l3);
    uint2 hp = { (unsigned)h0 | ((unsigned)h1 << 16), (unsigned)h2 | ((unsigned)h3 << 16) };
    uint2 lp = { (unsigned)l0 | ((unsigned)l1 << 16), (unsigned)l2 | ((unsigned)l3 << 16) };
    ((uint2*)hi)[i] = hp;
    ((uint2*)lo)[i] = lp;
}

// ---------------------------------------------------------------------------
// Prep kernel 2: tiled transpose + split: in (E,R,C) fp32 -> out (E,C,R) bf16.
// ---------------------------------------------------------------------------
__global__ __launch_bounds__(256)
void transpose_split_kernel(const float* __restrict__ in,
                            unsigned short* __restrict__ hiT,
                            unsigned short* __restrict__ loT,
                            int R, int C) {
    __shared__ float t[32][33];
    const int e  = blockIdx.z;
    const int c0 = blockIdx.x * 32, r0 = blockIdx.y * 32;
    const int tx = threadIdx.x, ty = threadIdx.y;
    const float* src = in + (size_t)e * R * C;
#pragma unroll
    for (int i = ty; i < 32; i += 8)
        t[i][tx] = src[(size_t)(r0 + i) * C + c0 + tx];
    __syncthreads();
    const size_t obase = (size_t)e * R * C;
#pragma unroll
    for (int i = ty; i < 32; i += 8) {
        unsigned short h, l;
        split_bf16(t[tx][i], h, l);
        size_t o = obase + (size_t)(c0 + i) * R + r0 + tx;
        hiT[o] = h;
        loT[o] = l;
    }
}

// ---------------------------------------------------------------------------
// Main GEMM: C = act(A @ B^T-major + bias) with split-bf16 WMMA.
//   A planes: (Bt, M, K) row-major.   B planes: (E, N, K) row-major (N-major).
//   GELU=true : write gelu result as bf16 hi/lo planes (intermediate H)
//   GELU=false: write fp32, scaled by expert weight
// ---------------------------------------------------------------------------
template <bool GELU>
__global__ __launch_bounds__(256)
void ffn_gemm_kernel(const unsigned short* __restrict__ Ah,
                     const unsigned short* __restrict__ Al,
                     const unsigned short* __restrict__ Bth,
                     const unsigned short* __restrict__ Btl,
                     const float* __restrict__ biasB,   // (E, N)
                     const int*   __restrict__ expIdx,  // (B, 1)
                     const float* __restrict__ expW,    // (B, 1)
                     float*          __restrict__ OutF, // (B, M, N) fp32
                     unsigned short* __restrict__ OutHh,// (B, M, N) bf16 hi
                     unsigned short* __restrict__ OutHl,// (B, M, N) bf16 lo
                     int M, int N, int K)
{
    const int b = blockIdx.z;
    const int e = expIdx[b];
    const float scale = GELU ? 1.0f : expW[b];

    const unsigned short* __restrict__ Abh = Ah  + (size_t)b * M * K;
    const unsigned short* __restrict__ Abl = Al  + (size_t)b * M * K;
    const unsigned short* __restrict__ Bbh = Bth + (size_t)e * N * K;
    const unsigned short* __restrict__ Bbl = Btl + (size_t)e * N * K;
    const float* __restrict__ biasP = biasB + (size_t)e * N;

    const int tile_n0 = blockIdx.x * TILE_N;
    const int tile_m0 = blockIdx.y * TILE_M;

    __shared__ __align__(16) unsigned short sAh[2][TILE_M * LDS_PAD];
    __shared__ __align__(16) unsigned short sAl[2][TILE_M * LDS_PAD];
    __shared__ __align__(16) unsigned short sBh[2][TILE_N * LDS_PAD];
    __shared__ __align__(16) unsigned short sBl[2][TILE_N * LDS_PAD];

    const int tid   = threadIdx.x;
    const int wave  = tid >> 5;
    const int lane  = tid & 31;
    const int waveM = wave & 3;          // 4 waves over M (32 rows each)
    const int waveN = wave >> 2;         // 2 waves over N (64 cols each)
    const int lr    = lane & 15;
    const int khalf = (lane >> 4) << 3;  // 0 or 8

    // --- per-lane async-copy slice: 512 x 16B chunks per plane per stage,
    //     each lane copies 2 chunks per plane (8 async issues / wave / stage)
    const int c0  = wave * 64 + lane;
    const int c1  = c0 + 32;
    const int r0_ = c0 >> 2, o0_ = (c0 & 3) * 8;   // row 0..127, elem offset
    const int r1_ = c1 >> 2, o1_ = (c1 & 3) * 8;

    const unsigned uAh0 = lds_off(&sAh[0][r0_ * LDS_PAD + o0_]);
    const unsigned uAh1 = lds_off(&sAh[0][r1_ * LDS_PAD + o1_]);
    const unsigned uAl0 = lds_off(&sAl[0][r0_ * LDS_PAD + o0_]);
    const unsigned uAl1 = lds_off(&sAl[0][r1_ * LDS_PAD + o1_]);
    const unsigned uBh0 = lds_off(&sBh[0][r0_ * LDS_PAD + o0_]);
    const unsigned uBh1 = lds_off(&sBh[0][r1_ * LDS_PAD + o1_]);
    const unsigned uBl0 = lds_off(&sBl[0][r0_ * LDS_PAD + o0_]);
    const unsigned uBl1 = lds_off(&sBl[0][r1_ * LDS_PAD + o1_]);

    const size_t gA0 = (size_t)(tile_m0 + r0_) * K + o0_;
    const size_t gA1 = (size_t)(tile_m0 + r1_) * K + o1_;
    const size_t gB0 = (size_t)(tile_n0 + r0_) * K + o0_;
    const size_t gB1 = (size_t)(tile_n0 + r1_) * K + o1_;

    auto issue_stage = [&](int kt, int buf) {
        const unsigned bo = (unsigned)(buf * STAGE_BYTES);
        const int k0 = kt * TILE_K;
        async_b128(uAh0 + bo, Abh + gA0 + k0);
        async_b128(uAh1 + bo, Abh + gA1 + k0);
        async_b128(uAl0 + bo, Abl + gA0 + k0);
        async_b128(uAl1 + bo, Abl + gA1 + k0);
        async_b128(uBh0 + bo, Bbh + gB0 + k0);
        async_b128(uBh1 + bo, Bbh + gB1 + k0);
        async_b128(uBl0 + bo, Bbl + gB0 + k0);
        async_b128(uBl1 + bo, Bbl + gB1 + k0);
    };

    v8f acc[2][4] = {};                  // 32x64 fp32 per wave

    issue_stage(0, 0);
    const int KT = K / TILE_K;
    int buf = 0;
#pragma unroll 1
    for (int kt = 0; kt < KT; ++kt) {
        wait_async0();                   // own stage-kt copies done
        __syncthreads();                 // everyone's copies visible
        if (kt + 1 < KT) issue_stage(kt + 1, buf ^ 1);  // overlap w/ compute

        v16bf a_h[2], a_l[2];
#pragma unroll
        for (int mi = 0; mi < 2; ++mi) {
            int r = waveM * 32 + mi * 16 + lr;
            a_h[mi] = load_frag(sAh[buf], r, khalf);
            a_l[mi] = load_frag(sAl[buf], r, khalf);
        }
#pragma unroll
        for (int ni = 0; ni < 4; ++ni) {
            int r = waveN * 64 + ni * 16 + lr;
            v16bf b_h = load_frag(sBh[buf], r, khalf);
            v16bf b_l = load_frag(sBl[buf], r, khalf);
#pragma unroll
            for (int mi = 0; mi < 2; ++mi) {
                acc[mi][ni] = __builtin_amdgcn_wmma_f32_16x16x32_bf16(
                    false, a_h[mi], false, b_h, (short)0, acc[mi][ni], false, false);
                acc[mi][ni] = __builtin_amdgcn_wmma_f32_16x16x32_bf16(
                    false, a_h[mi], false, b_l, (short)0, acc[mi][ni], false, false);
                acc[mi][ni] = __builtin_amdgcn_wmma_f32_16x16x32_bf16(
                    false, a_l[mi], false, b_h, (short)0, acc[mi][ni], false, false);
            }
        }
        __syncthreads();                 // buf safe to overwrite next round
        buf ^= 1;
    }

    // ---- epilogue. C layout (ISA 7.12.2): lanes 0-15: N=lane, M=vgpr;
    //      lanes 16-31: N=lane-16, M=vgpr+8.
    const int mHalf = (lane >> 4) << 3;
#pragma unroll
    for (int ni = 0; ni < 4; ++ni) {
        int nc = tile_n0 + waveN * 64 + ni * 16 + lr;
        float bias = biasP[nc];
#pragma unroll
        for (int mi = 0; mi < 2; ++mi) {
            int mBase = tile_m0 + waveM * 32 + mi * 16 + mHalf;
            v8f a = acc[mi][ni];
#pragma unroll
            for (int v = 0; v < 8; ++v) {
                float val = a[v] + bias;
                size_t off = (size_t)b * M * N + (size_t)(mBase + v) * N + nc;
                if (GELU) {
                    val = 0.5f * val * (1.0f + erff(val * 0.70710678118654752f));
                    unsigned short h, l;
                    split_bf16(val, h, l);
                    OutHh[off] = h;
                    OutHl[off] = l;
                } else {
                    OutF[off] = val * scale;
                }
            }
        }
    }
}

// ---------------------------------------------------------------------------

extern "C" void kernel_launch(void* const* d_in, const int* in_sizes, int n_in,
                              void* d_out, int out_size, void* d_ws, size_t ws_size,
                              hipStream_t stream) {
    (void)in_sizes; (void)n_in; (void)out_size; (void)ws_size;

    constexpr int B = 8, S = 4096, D = 1024, F = 4 * D, E = 8;

    const float* x   = (const float*)d_in[0];
    const float* ew  = (const float*)d_in[1];
    const float* W1  = (const float*)d_in[2];
    const float* b1  = (const float*)d_in[3];
    const float* W2  = (const float*)d_in[4];
    const float* b2  = (const float*)d_in[5];
    const int*   idx = (const int*)  d_in[6];

    constexpr size_t XN  = (size_t)B * S * D;   // x elements
    constexpr size_t W1N = (size_t)E * D * F;   // W1 elements
    constexpr size_t W2N = (size_t)E * F * D;   // W2 elements
    constexpr size_t HN  = (size_t)B * S * F;   // H elements

    unsigned short* xh  = (unsigned short*)d_ws;
    unsigned short* xl  = xh  + XN;
    unsigned short* w1h = xl  + XN;
    unsigned short* w1l = w1h + W1N;
    unsigned short* w2h = w1l + W1N;
    unsigned short* w2l = w2h + W2N;
    unsigned short* Hh  = w2l + W2N;
    unsigned short* Hl  = Hh  + HN;

    float* out = (float*)d_out;

    // ---- prep: split x; transpose+split W1 (E,D,F)->(E,F,D), W2 (E,F,D)->(E,D,F)
    {
        size_t n4 = XN / 4;
        split_planes_kernel<<<(unsigned)((n4 + 255) / 256), 256, 0, stream>>>(x, xh, xl, n4);
        dim3 tb(32, 8);
        transpose_split_kernel<<<dim3(F / 32, D / 32, E), tb, 0, stream>>>(W1, w1h, w1l, D, F);
        transpose_split_kernel<<<dim3(D / 32, F / 32, E), tb, 0, stream>>>(W2, w2h, w2l, F, D);
    }

    dim3 blk(256);
    dim3 g1(F / TILE_N, S / TILE_M, B);
    dim3 g2(D / TILE_N, S / TILE_M, B);

    // H = gelu(X @ W1[e] + b1[e])  -> bf16 hi/lo planes
    ffn_gemm_kernel<true ><<<g1, blk, 0, stream>>>(xh, xl, w1h, w1l, b1, idx, ew,
                                                   nullptr, Hh, Hl, S, F, D);
    // Y = (H @ W2[e] + b2[e]) * w[b]
    ffn_gemm_kernel<false><<<g2, blk, 0, stream>>>(Hh, Hl, w2h, w2l, b2, idx, ew,
                                                   out, nullptr, nullptr, S, D, F);
}
